// TextCrossAttention_87711822119004
// MI455X (gfx1250) — compile-verified, matrix operand order checked by reading, not run on hardware
//
#include <hip/hip_runtime.h>
#include <hip/hip_bf16.h>
#include <cstdint>
#include <cstddef>

// Set to 0 if the tensor_load_to_lds inline-asm form is rejected by the MC layer.
#define USE_TDM 1

// ---------------- problem constants ----------------
#define B_SZ    16
#define C_T     4
#define H_TOK   96          // tokens per sequence (H in reference)
#define N_ND    2000
#define DM      384         // d_model
#define K_N     8
#define K_QT    4
#define N_HEADS 8
#define HD      48
#define D_FF    1536
#define SEQS    (B_SZ * C_T * K_N)    // 512
#define MROWS   (SEQS * H_TOK)        // 49152 rows for all GEMMs
static const size_t MD = (size_t)MROWS * DM;   // 18,874,368

typedef __attribute__((ext_vector_type(8)))  float    v8f;
typedef __attribute__((ext_vector_type(8)))  __bf16   v8bf;
typedef __attribute__((ext_vector_type(16))) __bf16   v16bf;
typedef __attribute__((ext_vector_type(4)))  uint32_t u32x4;
typedef __attribute__((ext_vector_type(8)))  uint32_t u32x8;

// ---------------- small device helpers ----------------
__device__ __forceinline__ v8f vzero8() {
  v8f z;
#pragma unroll
  for (int i = 0; i < 8; ++i) z[i] = 0.0f;
  return z;
}
__device__ __forceinline__ v8bf bzero8() {
  v8bf z;
#pragma unroll
  for (int i = 0; i < 8; ++i) z[i] = (__bf16)0.0f;
  return z;
}

__device__ __forceinline__ v8f wmma_bf16(v16bf a, v16bf b, v8f c) {
  // D = A(16x32) * B(32x16) + C, f32 accumulate
  return __builtin_amdgcn_wmma_f32_16x16x32_bf16(false, a, false, b, (short)0, c,
                                                 false, false);
}

__device__ __forceinline__ void wait_tensorcnt0() {
#if defined(__has_builtin) && __has_builtin(__builtin_amdgcn_s_wait_tensorcnt)
  __builtin_amdgcn_s_wait_tensorcnt(0);
#else
  asm volatile("s_wait_tensorcnt 0x0" ::: "memory");
#endif
}

// Tensor Data Mover: DMA a tile_y x tile_x (elements, 2 B each) tile of a
// row-major matrix (leading dim `ld` elements) into LDS at lds_byte_off,
// packed contiguously [tile_y][tile_x]. D# per CDNA5 ISA ch.8 (2D: groups 0,1
// only; VADDR2/3 NULL). Tracked by TENSORcnt.
__device__ __forceinline__ void tdm_load_2d_bf16(const void* gaddr,
                                                 uint32_t lds_byte_off,
                                                 uint32_t tile_x, uint32_t tile_y,
                                                 uint32_t ld, uint32_t dim0,
                                                 uint32_t dim1) {
#if USE_TDM
  uint64_t ga = (uint64_t)(uintptr_t)gaddr;
  u32x4 g0;
  g0[0] = 1u;                                         // count=1 (valid), user mode
  g0[1] = lds_byte_off;                               // lds_addr
  g0[2] = (uint32_t)ga;                               // global_addr[31:0]
  g0[3] = (uint32_t)((ga >> 32) & 0x01FFFFFFu) | (2u << 30);  // addr[56:32]|type=2
  u32x8 g1;
  g1[0] = 1u << 16;                                   // data_size=1 (2 bytes)
  g1[1] = (dim0 & 0xFFFFu) << 16;                     // tensor_dim0[15:0]
  g1[2] = (dim0 >> 16) | ((dim1 & 0xFFFFu) << 16);    // dim0[31:16] | dim1[15:0]
  g1[3] = (dim1 >> 16) | (tile_x << 16);              // dim1[31:16] | tile_dim0
  g1[4] = tile_y & 0xFFFFu;                           // tile_dim1 (tile_dim2=0)
  g1[5] = ld;                                         // tensor_dim0_stride[31:0]
  g1[6] = 0u;                                         // stride[47:32]=0, d1s[15:0]=0
  g1[7] = 0u;
  asm volatile("tensor_load_to_lds %0, %1" ::"s"(g0), "s"(g1) : "memory");
#else
  (void)gaddr; (void)lds_byte_off; (void)tile_x; (void)tile_y; (void)ld;
  (void)dim0; (void)dim1;
#endif
}

// A/B fragment loader, row-major [rows x ld] bf16, 16-bit 16x32 WMMA layout:
// lane L: row = row0 + (L&15); halves j<8 -> k = k0 + (L>=16?8:0) + j
//         halves j>=8 -> k = k0 + 16 + (L>=16?8:0) + (j-8)
__device__ __forceinline__ v16bf frag_rm(const __bf16* __restrict__ base, int ld,
                                         int row0, int k0, int lane) {
  int r  = row0 + (lane & 15);
  int kb = k0 + ((lane & 16) ? 8 : 0);
  const __bf16* p = base + (size_t)r * ld + kb;
  v8bf lo = *(const v8bf*)p;
  v8bf hi = *(const v8bf*)(p + 16);
  v16bf f;
#pragma unroll
  for (int i = 0; i < 8; ++i) { f[i] = lo[i]; f[i + 8] = hi[i]; }
  return f;
}

// Same but for K-extent 48 (hd): k>=48 reads return zero (zero-padded tail).
__device__ __forceinline__ v16bf frag_rm48(const __bf16* __restrict__ base,
                                           int row0, int k0, int lane) {
  int r  = row0 + (lane & 15);
  int kb = k0 + ((lane & 16) ? 8 : 0);
  const __bf16* p = base + (size_t)r * 48 + kb;
  v8bf lo = (kb < 48) ? *(const v8bf*)p : bzero8();
  v8bf hi = (kb + 16 < 48) ? *(const v8bf*)(p + 16) : bzero8();
  v16bf f;
#pragma unroll
  for (int i = 0; i < 8; ++i) { f[i] = lo[i]; f[i + 8] = hi[i]; }
  return f;
}

// ---------------- elementwise kernels ----------------
__global__ void f32_to_bf16_kernel(const float* __restrict__ a,
                                   __bf16* __restrict__ o, size_t n) {
  size_t i = (size_t)blockIdx.x * blockDim.x + threadIdx.x;
  size_t stride = (size_t)gridDim.x * blockDim.x;
  for (; i < n; i += stride) o[i] = (__bf16)a[i];
}

__global__ void copy_f32_kernel(const float* __restrict__ a,
                                float* __restrict__ o, size_t n) {
  size_t i = (size_t)blockIdx.x * blockDim.x + threadIdx.x;
  size_t stride = (size_t)gridDim.x * blockDim.x;
  for (; i < n; i += stride) o[i] = a[i];
}

// ---------------- top-k retrieval ----------------
__device__ __forceinline__ void topk4_insert(float v, int id, float* bv, int* bi) {
  if (v > bv[3]) {
    bv[3] = v; bi[3] = id;
#pragma unroll
    for (int j = 3; j > 0; --j) {
      if (bv[j] > bv[j - 1]) {
        float tv = bv[j]; bv[j] = bv[j - 1]; bv[j - 1] = tv;
        int   ti = bi[j]; bi[j] = bi[j - 1]; bi[j - 1] = ti;
      }
    }
  }
}

// one block per query row; picks top-4 node indices
__global__ __launch_bounds__(256) void topk_kernel(
    const float* __restrict__ q, const float* __restrict__ nd,
    int* __restrict__ outIdx, int rowsPerBatch) {
  int bid = blockIdx.x;
  int b = bid / rowsPerBatch;
  __shared__ float qs[DM];
  __shared__ float cv[1024];
  __shared__ int   ci[1024];
  for (int i = threadIdx.x; i < DM; i += 256) qs[i] = q[(size_t)bid * DM + i];
  __syncthreads();

  float bv[4] = {-3.0e38f, -3.0e38f, -3.0e38f, -3.0e38f};
  int   bi[4] = {-1, -1, -1, -1};
  const float* ndb = nd + (size_t)b * N_ND * DM;
  for (int n = threadIdx.x; n < N_ND; n += 256) {
    const float* row = ndb + (size_t)n * DM;
    float dot = 0.0f;
#pragma unroll 4
    for (int k = 0; k < DM; ++k) dot += qs[k] * row[k];
    topk4_insert(dot, n, bv, bi);
  }
#pragma unroll
  for (int j = 0; j < 4; ++j) {
    cv[threadIdx.x * 4 + j] = bv[j];
    ci[threadIdx.x * 4 + j] = bi[j];
  }
  __syncthreads();
  if (threadIdx.x == 0) {
    float tv[4] = {-3.0e38f, -3.0e38f, -3.0e38f, -3.0e38f};
    int   ti[4] = {0, 0, 0, 0};
    for (int i = 0; i < 1024; ++i)
      if (ci[i] >= 0) topk4_insert(cv[i], ci[i], tv, ti);
#pragma unroll
    for (int j = 0; j < 4; ++j) outIdx[bid * 4 + j] = ti[j];
  }
}

// ---------------- gather / tile builders ----------------
__global__ void build_ref_kernel(const float* __restrict__ nd,
                                 const int* __restrict__ idx_qt,
                                 const int* __restrict__ idx_des,
                                 float* __restrict__ ref) {
  int row = blockIdx.x;            // 0 .. SEQS*H_TOK-1
  int s = row / H_TOK, h = row % H_TOK;
  int j = s % K_N;
  int bc = s / K_N;                // b*C_T + c
  int b = bc / C_T;
  int node = (j < K_QT) ? idx_qt[(bc * H_TOK + h) * 4 + j]
                        : idx_des[bc * 4 + (j - K_QT)];
  const float* src = nd + ((size_t)b * N_ND + node) * DM;
  float* dst = ref + (size_t)row * DM;
  for (int i = threadIdx.x; i < DM; i += blockDim.x) dst[i] = src[i];
}

// mode 0: src = des_emb (B,C_T,1,D)  broadcast over j,h
// mode 1: src = qt_emb  (B,C_T,H,D)  broadcast over j
__global__ void build_tile_kernel(const float* __restrict__ src,
                                  float* __restrict__ dst, int mode) {
  int row = blockIdx.x;
  int s = row / H_TOK, h = row % H_TOK;
  int bc = s / K_N;
  const float* sp = (mode == 0) ? src + (size_t)bc * DM
                                : src + ((size_t)bc * H_TOK + h) * DM;
  float* dp = dst + (size_t)row * DM;
  for (int i = threadIdx.x; i < DM; i += blockDim.x) dp[i] = sp[i];
}

// ---------------- LayerNorm (row = 384) ----------------
template <bool BF16OUT>
__global__ __launch_bounds__(128) void layernorm_kernel(
    const float* __restrict__ x, const float* __restrict__ g,
    const float* __restrict__ bta, void* __restrict__ outv) {
  int row = blockIdx.x, tid = threadIdx.x;
  const float* xr = x + (size_t)row * DM;
  float v0 = xr[tid], v1 = xr[tid + 128], v2 = xr[tid + 256];
  __shared__ float red[128];
  red[tid] = v0 + v1 + v2;
  __syncthreads();
  for (int o = 64; o > 0; o >>= 1) {
    if (tid < o) red[tid] += red[tid + o];
    __syncthreads();
  }
  float mean = red[0] * (1.0f / DM);
  __syncthreads();
  float d0 = v0 - mean, d1 = v1 - mean, d2 = v2 - mean;
  red[tid] = d0 * d0 + d1 * d1 + d2 * d2;
  __syncthreads();
  for (int o = 64; o > 0; o >>= 1) {
    if (tid < o) red[tid] += red[tid + o];
    __syncthreads();
  }
  float rstd = rsqrtf(red[0] * (1.0f / DM) + 1e-5f);
  float y0 = d0 * rstd * g[tid] + bta[tid];
  float y1 = d1 * rstd * g[tid + 128] + bta[tid + 128];
  float y2 = d2 * rstd * g[tid + 256] + bta[tid + 256];
  if (BF16OUT) {
    __bf16* o = (__bf16*)outv + (size_t)row * DM;
    o[tid] = (__bf16)y0; o[tid + 128] = (__bf16)y1; o[tid + 256] = (__bf16)y2;
  } else {
    float* o = (float*)outv + (size_t)row * DM;
    o[tid] = y0; o[tid + 128] = y1; o[tid + 256] = y2;
  }
}

// ---------------- WMMA GEMM: C[M,N] = A[M,K] * W[N,K]^T + bias (+res | gelu) ---
// EPI 0: bias, f32 out.  EPI 1: bias + residual, f32 out.  EPI 2: bias + exact
// GELU, bf16 out.  Block = 256 threads = 8 waves (4x2 tile grid); workgroup
// tile 128(M) x 64(N); each wave owns 32x32 via 4 WMMA accumulators.
// K-loop double-buffers LDS: A tile 128x32 staged via b128 loads + ds_store,
// W tile 64x32 DMA'd by the Tensor Data Mover (wave 0), TENSORcnt-synced.
template <int EPI>
__global__ __launch_bounds__(256) void gemm_wmma_kernel(
    const __bf16* __restrict__ A, int lda, const __bf16* __restrict__ W, int ldw,
    const float* __restrict__ bias, const float* __restrict__ resid, int ldr,
    float* __restrict__ outF, __bf16* __restrict__ outB, int ldc, int Ktot) {
  __shared__ __align__(16) __bf16 At[2][128 * 32];
  __shared__ __align__(16) __bf16 Bt[2][64 * 32];

  int tid  = threadIdx.x;
  int lane = tid & 31;
  int wave = tid >> 5;
  int m0 = (wave & 3) * 32;            // within 128-row tile
  int n0 = (wave >> 2) * 32;           // within 64-col tile
  int mblk = blockIdx.y * 128;
  int nblk = blockIdx.x * 64;
  int Ntot = gridDim.x * 64;

  // A-tile staging: 128x32 bf16 = 512 16-byte chunks, 2 per thread
  int arow0 = tid >> 2, acol0 = (tid & 3) * 8;          // chunk tid
  int arow1 = (tid + 256) >> 2, acol1 = ((tid + 256) & 3) * 8;
  v8bf areg0, areg1;
  auto loadA = [&](int k0) {
    areg0 = *(const v8bf*)(A + (size_t)(mblk + arow0) * lda + k0 + acol0);
    areg1 = *(const v8bf*)(A + (size_t)(mblk + arow1) * lda + k0 + acol1);
  };
  auto storeA = [&](int buf) {
    *(v8bf*)(&At[buf][arow0 * 32 + acol0]) = areg0;
    *(v8bf*)(&At[buf][arow1 * 32 + acol1]) = areg1;
  };
#if USE_TDM
  auto issueB = [&](int t, int buf) {
    if (wave == 0)
      tdm_load_2d_bf16(W + (size_t)nblk * ldw + t * 32,
                       (uint32_t)(uintptr_t)(void*)&Bt[buf][0],
                       /*tile_x=*/32, /*tile_y=*/64, /*ld=*/(uint32_t)ldw,
                       /*dim0=*/(uint32_t)(ldw - t * 32),
                       /*dim1=*/(uint32_t)(Ntot - nblk));
  };
  auto waitB = [&]() { if (wave == 0) wait_tensorcnt0(); };
#else
  int brow0 = tid >> 1, bcol0 = (tid & 1) * 16;         // 64x32 = 128 chunks x32B
  v8bf breg0, breg1;
  auto issueB = [&](int t, int buf) {
    (void)buf;
    if (tid < 128) {
      breg0 = *(const v8bf*)(W + (size_t)(nblk + brow0) * ldw + t * 32 + bcol0);
      breg1 = *(const v8bf*)(W + (size_t)(nblk + brow0) * ldw + t * 32 + bcol0 + 8);
    }
  };
  auto waitB = [&](int buf) {};
#endif

  loadA(0);
  issueB(0, 0);
  storeA(0);
#if !USE_TDM
  if (tid < 128) {
    *(v8bf*)(&Bt[0][brow0 * 32 + bcol0]) = breg0;
    *(v8bf*)(&Bt[0][brow0 * 32 + bcol0 + 8]) = breg1;
  }
#endif
  waitB();
  __syncthreads();

  v8f acc00 = vzero8(), acc01 = vzero8(), acc10 = vzero8(), acc11 = vzero8();
  int T = Ktot >> 5;
  for (int t = 0; t < T; ++t) {
    int cur = t & 1, nxt = cur ^ 1;
    if (t + 1 < T) {
      loadA((t + 1) * 32);
      issueB(t + 1, nxt);
    }
    v16bf a0 = frag_rm(&At[cur][0], 32, m0, 0, lane);
    v16bf a1 = frag_rm(&At[cur][0], 32, m0 + 16, 0, lane);
    v16bf b0 = frag_rm(&Bt[cur][0], 32, n0, 0, lane);
    v16bf b1 = frag_rm(&Bt[cur][0], 32, n0 + 16, 0, lane);
    acc00 = wmma_bf16(a0, b0, acc00);
    acc01 = wmma_bf16(a0, b1, acc01);
    acc10 = wmma_bf16(a1, b0, acc10);
    acc11 = wmma_bf16(a1, b1, acc11);
    if (t + 1 < T) {
      storeA(nxt);
#if !USE_TDM
      if (tid < 128) {
        *(v8bf*)(&Bt[nxt][brow0 * 32 + bcol0]) = breg0;
        *(v8bf*)(&Bt[nxt][brow0 * 32 + bcol0 + 8]) = breg1;
      }
#endif
      waitB();
      __syncthreads();
    }
  }

  int mh = (lane & 16) ? 8 : 0;
  int nl = lane & 15;
#pragma unroll
  for (int t = 0; t < 4; ++t) {
    v8f acc = (t == 0) ? acc00 : (t == 1) ? acc01 : (t == 2) ? acc10 : acc11;
    int mb = mblk + m0 + (t >> 1) * 16 + mh;
    int nn = nblk + n0 + (t & 1) * 16 + nl;
    float bv = bias[nn];
#pragma unroll
    for (int r = 0; r < 8; ++r) {
      size_t idx = (size_t)(mb + r) * ldc + nn;
      float v = acc[r] + bv;
      if (EPI == 1) v += resid[(size_t)(mb + r) * ldr + nn];
      if (EPI == 2) {
        v = 0.5f * v * (1.0f + erff(v * 0.70710678118f));
        outB[idx] = (__bf16)v;
      } else {
        outF[idx] = v;
      }
    }
  }
}

// ---------------- fused attention (one block per sequence*head) -------------
// qkv layout: [SEQS*96, 1152] f32; q | k | v each 384 wide, head h at col h*48.
// LDS phases:  qs[96x48] ks[96x48] vt[48x96] (bf16) + sc[96x96] f32 = 63 KB;
// pb[96x96] bf16 overlays qs+ks after the score barrier.
__global__ __launch_bounds__(192) void attention_kernel(
    const float* __restrict__ qkv, __bf16* __restrict__ outB) {
  int s = blockIdx.x >> 3;
  int h = blockIdx.x & 7;
  int tid = threadIdx.x;
  int lane = tid & 31, wave = tid >> 5;

  __shared__ __align__(16) unsigned char smem[64512];
  __bf16* qs = (__bf16*)smem;              // 9216 B
  __bf16* ks = qs + 96 * 48;               // 9216 B
  __bf16* vt = ks + 96 * 48;               // [48][96] transposed, 9216 B
  float*  sc = (float*)(smem + 27648);     // [96][96] 36864 B
  __bf16* pb = (__bf16*)smem;              // [96][96] overlays qs..ks

  const float* base = qkv + (size_t)s * 96 * 1152;
  const float scale = 0.1443375673f;       // 1/sqrt(48)
  for (int idx = tid; idx < 96 * 48; idx += 192) {
    int t = idx / 48, c = idx % 48;
    const float* rp = base + (size_t)t * 1152 + h * 48 + c;
    qs[idx] = (__bf16)(rp[0] * scale);
    ks[idx] = (__bf16)(rp[384]);
    vt[c * 96 + t] = (__bf16)(rp[768]);
  }
  __syncthreads();

  // scores: S = q * k^T  (96x96), wave w owns row strip [16w, 16w+16)
  int m0 = wave * 16;
  v8f sacc[6];
#pragma unroll
  for (int j = 0; j < 6; ++j) sacc[j] = vzero8();
  for (int kk = 0; kk < 64; kk += 32) {
    v16bf a = frag_rm48(qs, m0, kk, lane);
#pragma unroll
    for (int j = 0; j < 6; ++j) {
      v16bf bb = frag_rm48(ks, j * 16, kk, lane);
      sacc[j] = wmma_bf16(a, bb, sacc[j]);
    }
  }
  int mh = (lane & 16) ? 8 : 0;
  int nl = lane & 15;
#pragma unroll
  for (int j = 0; j < 6; ++j)
#pragma unroll
    for (int r = 0; r < 8; ++r)
      sc[(m0 + mh + r) * 96 + (j * 16 + nl)] = sacc[j][r];
  __syncthreads();

  // softmax (f32) -> probabilities as bf16 in pb
  if (tid < 96) {
    float* rowp = sc + tid * 96;
    float mx = rowp[0];
    for (int i = 1; i < 96; ++i) mx = fmaxf(mx, rowp[i]);
    float sum = 0.0f;
    for (int i = 0; i < 96; ++i) {
      float e = __expf(rowp[i] - mx);
      sum += e;
      rowp[i] = e;
    }
    float inv = 1.0f / sum;
    for (int i = 0; i < 96; ++i) pb[tid * 96 + i] = (__bf16)(rowp[i] * inv);
  }
  __syncthreads();

  // O = P(96x96) * V(96x48); vt holds V^T so B[k][n] = vt[n][k]
  v8f oacc[3];
#pragma unroll
  for (int j = 0; j < 3; ++j) oacc[j] = vzero8();
  for (int kk = 0; kk < 96; kk += 32) {
    v16bf a = frag_rm(pb, 96, m0, kk, lane);
#pragma unroll
    for (int j = 0; j < 3; ++j) {
      v16bf bb = frag_rm(vt, 96, j * 16, kk, lane);
      oacc[j] = wmma_bf16(a, bb, oacc[j]);
    }
  }
#pragma unroll
  for (int j = 0; j < 3; ++j)
#pragma unroll
    for (int r = 0; r < 8; ++r) {
      int m = m0 + mh + r;
      int c = j * 16 + nl;
      outB[((size_t)s * 96 + m) * DM + h * 48 + c] = (__bf16)oacc[j][r];
    }
}

// ---------------- host orchestration ----------------
static inline size_t wbf_layer_off(int dec, int l) {
  return ((size_t)dec * 3 + l) * 2359296;  // bf16 elems per layer
}
// offsets within a layer chunk (bf16 elems)
#define OFF_SA_IN  0
#define OFF_SA_OUT 442368
#define OFF_CA_IN  589824
#define OFF_CA_OUT 1032192
#define OFF_FF1    1179648
#define OFF_FF2    1769472

extern "C" void kernel_launch(void* const* d_in, const int* in_sizes, int n_in,
                              void* d_out, int out_size, void* d_ws, size_t ws_size,
                              hipStream_t stream) {
  (void)in_sizes; (void)n_in; (void)out_size; (void)ws_size;

  // inputs in setup_inputs() dict (insertion) order:
  const float* qt_emb  = (const float*)d_in[0];
  const float* des_emb = (const float*)d_in[1];
  const float* nd_emb  = (const float*)d_in[2];
  const int PB = 3;  // params base; per layer (insertion order):
  // 0 sa_in_w 1 sa_in_b 2 sa_out_w 3 sa_out_b 4 ca_in_w 5 ca_in_b
  // 6 ca_out_w 7 ca_out_b 8 ff1_w 9 ff1_b 10 ff2_w 11 ff2_b
  // 12 ln1_g 13 ln1_b 14 ln2_g 15 ln2_b 16 ln3_g 17 ln3_b
  auto LW = [&](int dec, int l, int o) -> const float* {
    return (const float*)d_in[PB + dec * 56 + l * 18 + o];
  };
  auto DN = [&](int dec, int o) -> const float* {  // 0=norm_g 1=norm_b
    return (const float*)d_in[PB + dec * 56 + 54 + o];
  };

  // workspace carve-up
  size_t off = 0;
  auto take = [&](size_t bytes) -> char* {
    char* p = (char*)d_ws + off;
    off = (off + bytes + 255) & ~(size_t)255;
    return p;
  };
  float*  ref     = (float*)take(MD * 4);            // also reused as r2
  float*  bufX    = (float*)take(MD * 4);
  float*  bufR1   = (float*)take(MD * 4);
  __bf16* h_bf    = (__bf16*)take(MD * 2);
  __bf16* mem_bf  = (__bf16*)take(MD * 2);
  float*  qkv     = (float*)take((size_t)MROWS * 1152 * 4);
  __bf16* ff1_bf  = (__bf16*)qkv;                    // overlay (disjoint in time)
  __bf16* attn_bf = (__bf16*)take(MD * 2);
  __bf16* wbf     = (__bf16*)take(21233664ull * 2);  // all weights, bf16
  int*    idx_qt  = (int*)take((size_t)B_SZ * C_T * H_TOK * 4 * 4);
  int*    idx_des = (int*)take((size_t)B_SZ * C_T * 4 * 4);

  auto conv = [&](const float* src, __bf16* dst, size_t n) {
    int blocks = (int)((n + 255) / 256);
    if (blocks > 65535) blocks = 65535;
    f32_to_bf16_kernel<<<blocks, 256, 0, stream>>>(src, dst, n);
  };

  // 0. convert all weights to bf16 (L2-resident afterwards)
  for (int dec = 0; dec < 3; ++dec)
    for (int l = 0; l < 3; ++l) {
      size_t wl = wbf_layer_off(dec, l);
      conv(LW(dec, l, 0),  wbf + wl + OFF_SA_IN,  (size_t)1152 * 384);
      conv(LW(dec, l, 2),  wbf + wl + OFF_SA_OUT, (size_t)384 * 384);
      conv(LW(dec, l, 4),  wbf + wl + OFF_CA_IN,  (size_t)1152 * 384);
      conv(LW(dec, l, 6),  wbf + wl + OFF_CA_OUT, (size_t)384 * 384);
      conv(LW(dec, l, 8),  wbf + wl + OFF_FF1,    (size_t)1536 * 384);
      conv(LW(dec, l, 10), wbf + wl + OFF_FF2,    (size_t)384 * 1536);
    }

  // 1. retrieval
  topk_kernel<<<B_SZ * C_T * H_TOK, 256, 0, stream>>>(qt_emb, nd_emb, idx_qt,
                                                      C_T * H_TOK);
  topk_kernel<<<B_SZ * C_T, 256, 0, stream>>>(des_emb, nd_emb, idx_des, C_T);
  build_ref_kernel<<<SEQS * H_TOK, 128, 0, stream>>>(nd_emb, idx_qt, idx_des, ref);

  const dim3 blk(256);
  auto gemm_grid = [](int N) { return dim3((unsigned)(N / 64), MROWS / 128, 1); };

  auto run_decoder = [&](int dec, float* x, const float* mem, float* outLn) {
    conv(mem, mem_bf, MD);
    for (int l = 0; l < 3; ++l) {
      size_t wl = wbf_layer_off(dec, l);
      const __bf16* w_sa_in  = wbf + wl + OFF_SA_IN;
      const __bf16* w_sa_out = wbf + wl + OFF_SA_OUT;
      const __bf16* w_ca_in  = wbf + wl + OFF_CA_IN;
      const __bf16* w_ca_out = wbf + wl + OFF_CA_OUT;
      const __bf16* w_ff1    = wbf + wl + OFF_FF1;
      const __bf16* w_ff2    = wbf + wl + OFF_FF2;

      // self-attention block
      layernorm_kernel<true><<<MROWS, 128, 0, stream>>>(x, LW(dec, l, 12),
                                                        LW(dec, l, 13), h_bf);
      gemm_wmma_kernel<0><<<gemm_grid(1152), blk, 0, stream>>>(
          h_bf, DM, w_sa_in, DM, LW(dec, l, 1), nullptr, 0, qkv, nullptr, 1152, DM);
      attention_kernel<<<SEQS * N_HEADS, 192, 0, stream>>>(qkv, attn_bf);
      gemm_wmma_kernel<1><<<gemm_grid(DM), blk, 0, stream>>>(
          attn_bf, DM, w_sa_out, DM, LW(dec, l, 3), x, DM, x, nullptr, DM, DM);

      // cross-attention block (q from h, k/v from mem)
      layernorm_kernel<true><<<MROWS, 128, 0, stream>>>(x, LW(dec, l, 14),
                                                        LW(dec, l, 15), h_bf);
      gemm_wmma_kernel<0><<<gemm_grid(DM), blk, 0, stream>>>(
          h_bf, DM, w_ca_in, DM, LW(dec, l, 5), nullptr, 0, qkv, nullptr, 1152, DM);
      gemm_wmma_kernel<0><<<gemm_grid(768), blk, 0, stream>>>(
          mem_bf, DM, w_ca_in + (size_t)384 * 384, DM, LW(dec, l, 5) + 384,
          nullptr, 0, qkv + 384, nullptr, 1152, DM);
      attention_kernel<<<SEQS * N_HEADS, 192, 0, stream>>>(qkv, attn_bf);
      gemm_wmma_kernel<1><<<gemm_grid(DM), blk, 0, stream>>>(
          attn_bf, DM, w_ca_out, DM, LW(dec, l, 7), x, DM, x, nullptr, DM, DM);

      // FFN block (exact-erf GELU fused, bf16 intermediate)
      layernorm_kernel<true><<<MROWS, 128, 0, stream>>>(x, LW(dec, l, 16),
                                                        LW(dec, l, 17), h_bf);
      gemm_wmma_kernel<2><<<gemm_grid(D_FF), blk, 0, stream>>>(
          h_bf, DM, w_ff1, DM, LW(dec, l, 9), nullptr, 0, nullptr, ff1_bf, D_FF,
          DM);
      gemm_wmma_kernel<1><<<gemm_grid(DM), blk, 0, stream>>>(
          ff1_bf, D_FF, w_ff2, D_FF, LW(dec, l, 11), x, DM, x, nullptr, DM, D_FF);
    }
    layernorm_kernel<false><<<MROWS, 128, 0, stream>>>(x, DN(dec, 0), DN(dec, 1),
                                                       outLn);
  };

  // 2. dec1: tgt = des tile, mem = ref
  build_tile_kernel<<<SEQS * H_TOK, 128, 0, stream>>>(des_emb, bufX, 0);
  run_decoder(0, bufX, ref, bufR1);

  // 3. dec2: tgt = qt tile, mem = r1 ; r2 reuses ref buffer
  build_tile_kernel<<<SEQS * H_TOK, 128, 0, stream>>>(qt_emb, bufX, 1);
  run_decoder(1, bufX, bufR1, ref);

  // 4. dec3: tgt = r2 (copy), mem = r2 ; final LN straight into d_out
  copy_f32_kernel<<<4096, 256, 0, stream>>>(ref, bufX, MD);
  run_decoder(2, bufX, ref, (float*)d_out);
}
